// RoughMatchingModel_84774064488612
// MI455X (gfx1250) — matrix operand order repaired and checked by reading, not compiled, not run on hardware
//
#include <hip/hip_runtime.h>
#include <hip/hip_bf16.h>
#include <math.h>

// ---------------------------------------------------------------------------
// Problem constants (match reference)
// ---------------------------------------------------------------------------
#define NS_   100000
#define NT_   100000
#define Mn_   512
#define Dc_   256
#define DLc_  128
#define Hh_   4
#define Cc_   256
#define Kk_   64
#define ITERS_ 50
#define POS_MARGIN_ 0.1f

typedef __attribute__((ext_vector_type(16))) _Float16 v16h;
typedef __attribute__((ext_vector_type(8)))  float    v8f;

// ---------------------------------------------------------------------------
// Generic batched WMMA GEMM (compile-time TA/TB specialization):
//   Y[b, o, n] = act( alpha * sum_i A(b,o,i) * B(b,i,n) + bias[o] + res[b,o,n] )
// A(b,o,i) = A[b*sA + (TA ? i*lda+o : o*lda+i)]
// B(b,i,n) = Bm[b*sB + (TB ? n*ldb+i : i*ldb+n)]
// O mult of 16, I mult of 32, N mult of 16; lda/ldb mult of 16 (alignment).
// blockDim = 128 (4 waves), each wave owns one 16x16 output tile.
// ---------------------------------------------------------------------------
template<int TA, int TB>
__global__ void k_wmma_gemm(const float* __restrict__ A, int lda, long sA,
                            const float* __restrict__ Bm, int ldb, long sB,
                            const float* __restrict__ bias,
                            const float* __restrict__ res, long sRes,
                            float* __restrict__ Y, long sY,
                            int O, int I, int N, float alpha, int relu)
{
    int wave = threadIdx.x >> 5;
    int lane = threadIdx.x & 31;
    int n0 = blockIdx.x * 64 + wave * 16;
    int o0 = blockIdx.y * 16;
    long b = blockIdx.z;
    if (n0 >= N || o0 >= O) return;

    const float* Ab = A  + b * sA;
    const float* Bb = Bm + b * sB;

    int half = lane >> 4;   // 0: lanes 0-15, 1: lanes 16-31
    int lp   = lane & 15;
    int row  = o0 + lp;     // A-matrix row owned by this lane
    int col  = n0 + lp;     // B/C/D column owned by this lane

    // A fragment (16x32), contiguous k order used by this code (B matches): lanes 0-15 hold K=kc+[0..15];
    // lanes 16-31 hold K=kc+[16..31] for row M=lp.
    auto loadA = [&](int kc) -> v16h {
        v16h af;
        if (TA == 0) {
            const float* p = &Ab[(long)row * lda + kc + half * 16];
            float4 a0 = *(const float4*)(p);
            float4 a1 = *(const float4*)(p + 4);
            float4 a2 = *(const float4*)(p + 8);
            float4 a3 = *(const float4*)(p + 12);
            af[0]=(_Float16)a0.x; af[1]=(_Float16)a0.y; af[2]=(_Float16)a0.z; af[3]=(_Float16)a0.w;
            af[4]=(_Float16)a1.x; af[5]=(_Float16)a1.y; af[6]=(_Float16)a1.z; af[7]=(_Float16)a1.w;
            af[8]=(_Float16)a2.x; af[9]=(_Float16)a2.y; af[10]=(_Float16)a2.z; af[11]=(_Float16)a2.w;
            af[12]=(_Float16)a3.x; af[13]=(_Float16)a3.y; af[14]=(_Float16)a3.z; af[15]=(_Float16)a3.w;
        } else {
            int k0 = kc + half * 16;
#pragma unroll
            for (int j = 0; j < 8; ++j) {
                af[j]     = (_Float16)Ab[(long)(k0 + j) * lda + row];
                af[8 + j] = (_Float16)Ab[(long)(k0 + 8 + j) * lda + row];
            }
        }
        return af;
    };
    // B fragment (32x16), same contiguous k order as A: lanes 0-15 hold K=kc+[0..15],
    // lanes 16-31 hold K=kc+[16..31] for column N=lp.
    auto loadB = [&](int kc) -> v16h {
        v16h bf;
        if (TB == 1) {
            const float* p = &Bb[(long)col * ldb + kc + half * 16];
            float4 b0 = *(const float4*)(p);
            float4 b1 = *(const float4*)(p + 4);
            float4 b2 = *(const float4*)(p + 8);
            float4 b3 = *(const float4*)(p + 12);
            bf[0]=(_Float16)b0.x; bf[1]=(_Float16)b0.y; bf[2]=(_Float16)b0.z; bf[3]=(_Float16)b0.w;
            bf[4]=(_Float16)b1.x; bf[5]=(_Float16)b1.y; bf[6]=(_Float16)b1.z; bf[7]=(_Float16)b1.w;
            bf[8]=(_Float16)b2.x; bf[9]=(_Float16)b2.y; bf[10]=(_Float16)b2.z; bf[11]=(_Float16)b2.w;
            bf[12]=(_Float16)b3.x; bf[13]=(_Float16)b3.y; bf[14]=(_Float16)b3.z; bf[15]=(_Float16)b3.w;
        } else {
            int k0 = kc + half * 16;
#pragma unroll
            for (int j = 0; j < 16; ++j)
                bf[j] = (_Float16)Bb[(long)(k0 + j) * ldb + col];
        }
        return bf;
    };

    v8f acc = {};
    int kc = 0;
    for (; kc + 64 <= I; kc += 64) {
        // speculative prefetch two chunks ahead (dropped silently if OOB)
        __builtin_prefetch(TA ? &Ab[(long)(kc + 128) * lda + row]
                              : &Ab[(long)row * lda + kc + 128], 0, 1);
        __builtin_prefetch(TB ? &Bb[(long)col * ldb + kc + 128]
                              : &Bb[(long)(kc + 128) * ldb + col], 0, 1);
        v16h a0 = loadA(kc),      b0 = loadB(kc);
        v16h a1 = loadA(kc + 32), b1 = loadB(kc + 32);
        acc = __builtin_amdgcn_wmma_f32_16x16x32_f16(false, a0, false, b0, (short)0, acc, false, false);
        acc = __builtin_amdgcn_wmma_f32_16x16x32_f16(false, a1, false, b1, (short)0, acc, false, false);
    }
    if (kc < I) {
        v16h a0 = loadA(kc), b0 = loadB(kc);
        acc = __builtin_amdgcn_wmma_f32_16x16x32_f16(false, a0, false, b0, (short)0, acc, false, false);
    }

    // C/D layout: VGPR i, lanes 0-15: M = i, lanes 16-31: M = i + 8
#pragma unroll
    for (int i = 0; i < 8; ++i) {
        int orow = o0 + half * 8 + i;
        float v = acc[i] * alpha;
        if (bias) v += bias[orow];
        if (res)  v += res[b * sRes + (long)orow * N + col];
        if (relu) v = fmaxf(v, 0.f);
        Y[b * sY + (long)orow * N + col] = v;
    }
}

// ---------------------------------------------------------------------------
// Softmax over rows: one wave per row of length Nm
// ---------------------------------------------------------------------------
__global__ void k_softmax_rows(float* __restrict__ att, int Nm)
{
    long row = blockIdx.x;
    float* p = att + row * (long)Nm;
    int lane = threadIdx.x;
    float mx = -3.4e38f;
    for (int j = lane; j < Nm; j += 32) mx = fmaxf(mx, p[j]);
    for (int s = 16; s; s >>= 1) mx = fmaxf(mx, __shfl_xor(mx, s));
    float sum = 0.f;
    for (int j = lane; j < Nm; j += 32) { float e = __expf(p[j] - mx); p[j] = e; sum += e; }
    for (int s = 16; s; s >>= 1) sum += __shfl_xor(sum, s);
    float inv = 1.f / sum;
    for (int j = lane; j < Nm; j += 32) p[j] *= inv;
}

// ---------------------------------------------------------------------------
// NN argmin: for each point find nearest of m nodes (nodes cached in LDS)
// ---------------------------------------------------------------------------
__global__ void k_nn_argmin(const float* __restrict__ pts, int n,
                            const float* __restrict__ nodes, int m,
                            int* __restrict__ ids)
{
    __shared__ float sn[Mn_ * 3];
    for (int j = threadIdx.x; j < m * 3; j += blockDim.x) sn[j] = nodes[j];
    __syncthreads();
    int i = blockIdx.x * blockDim.x + threadIdx.x;
    if (i >= n) return;
    float px = pts[i * 3], py = pts[i * 3 + 1], pz = pts[i * 3 + 2];
    float best = 3.4e38f; int bj = 0;
    for (int j = 0; j < m; ++j) {
        float dx = px - sn[j * 3], dy = py - sn[j * 3 + 1], dz = pz - sn[j * 3 + 2];
        float d = dx * dx + dy * dy + dz * dz;
        if (d < best) { best = d; bj = j; }
    }
    ids[i] = bj;
}

__global__ void k_zero_f32(float* __restrict__ p, long n)
{ long i = blockIdx.x * (long)blockDim.x + threadIdx.x; if (i < n) p[i] = 0.f; }

__global__ void k_zero_i32(int* __restrict__ p, long n)
{ long i = blockIdx.x * (long)blockDim.x + threadIdx.x; if (i < n) p[i] = 0; }

__global__ void k_build_patch(const int* __restrict__ ids, int n,
                              int* __restrict__ cnt, int* __restrict__ patch)
{
    int i = blockIdx.x * blockDim.x + threadIdx.x;
    if (i >= n) return;
    int m = ids[i];
    int r = atomicAdd(&cnt[m], 1);
    if (r < Kk_) patch[m * Kk_ + r] = i;
}

__global__ void k_patch_mask(const int* __restrict__ cnt,
                             int* __restrict__ patch, float* __restrict__ mask)
{
    int i = blockIdx.x * blockDim.x + threadIdx.x;
    if (i >= Mn_ * Kk_) return;
    int m = i >> 6, k = i & 63;
    int c = cnt[m]; if (c > Kk_) c = Kk_;
    if (k < c) mask[i] = 0.f;
    else { mask[i] = 1.f; patch[i] = 0; }
}

__global__ void k_make_sel(const int* __restrict__ patch, const float* __restrict__ mask,
                           const int* __restrict__ ncorr, int colid,
                           int* __restrict__ sel, float* __restrict__ mout)
{
    int i = blockIdx.x * blockDim.x + threadIdx.x;
    if (i >= Cc_ * Kk_) return;
    int c = i >> 6, k = i & 63;
    int node = ncorr[c * 2 + colid];
    sel[i]  = patch[node * Kk_ + k];
    mout[i] = mask[node * Kk_ + k];
}

__global__ void k_gather_pts(const float* __restrict__ pcd, const int* __restrict__ sel,
                             const float* __restrict__ rot, const float* __restrict__ trans,
                             int n, float* __restrict__ out)
{
    int i = blockIdx.x * blockDim.x + threadIdx.x;
    if (i >= n) return;
    int s = sel[i];
    float x = pcd[s * 3], y = pcd[s * 3 + 1], z = pcd[s * 3 + 2];
    if (rot) {
        float nx = rot[0] * x + rot[1] * y + rot[2] * z + trans[0];
        float ny = rot[3] * x + rot[4] * y + rot[5] * z + trans[1];
        float nz = rot[6] * x + rot[7] * y + rot[8] * z + trans[2];
        x = nx; y = ny; z = nz;
    }
    out[i * 3] = x; out[i * 3 + 1] = y; out[i * 3 + 2] = z;
}

// gather point features into (C, DL, K) layout
__global__ void k_gather_feats(const float* __restrict__ f, const int* __restrict__ sel,
                               float* __restrict__ out)
{
    long i = blockIdx.x * (long)blockDim.x + threadIdx.x;
    long total = (long)Cc_ * DLc_ * Kk_;
    if (i >= total) return;
    int k = (int)(i % Kk_);
    long r = i / Kk_;
    int d = (int)(r % DLc_);
    int c = (int)(r / DLc_);
    out[i] = f[(long)sel[c * Kk_ + k] * DLc_ + d];
}

__global__ void k_concat(const float* __restrict__ x, const float* __restrict__ m,
                         float* __restrict__ cat, int B, int Dm, int N)
{
    long i = blockIdx.x * (long)blockDim.x + threadIdx.x;
    long total = (long)B * 2 * Dm * N;
    if (i >= total) return;
    int n = (int)(i % N);
    long r = i / N;
    int d = (int)(r % (2 * Dm));
    int b = (int)(r / (2 * Dm));
    cat[i] = (d < Dm) ? x[((long)b * Dm + d) * N + n]
                      : m[((long)b * Dm + (d - Dm)) * N + n];
}

// ---------------------------------------------------------------------------
// Coarse OT (log_optimal_transport): Z is 513x513, M=N=512
// ---------------------------------------------------------------------------
__global__ void k_assemble_Z(const float* __restrict__ core, const float* __restrict__ alpha,
                             float* __restrict__ Z)
{
    long i = blockIdx.x * (long)blockDim.x + threadIdx.x;
    long total = (long)(Mn_ + 1) * (Mn_ + 1);
    if (i >= total) return;
    int j = (int)(i % (Mn_ + 1));
    int r = (int)(i / (Mn_ + 1));
    Z[i] = (r < Mn_ && j < Mn_) ? core[(long)r * Mn_ + j] : alpha[0];
}

__global__ void k_ot_row(const float* __restrict__ Z, float* __restrict__ u,
                         const float* __restrict__ v)
{
    int i = blockIdx.x;      // 0..512
    int lane = threadIdx.x;
    const float* zr = Z + (long)i * (Mn_ + 1);
    float mx = -3.4e38f;
    for (int j = lane; j <= Mn_; j += 32) mx = fmaxf(mx, zr[j] + v[j]);
    for (int s = 16; s; s >>= 1) mx = fmaxf(mx, __shfl_xor(mx, s));
    float sm = 0.f;
    for (int j = lane; j <= Mn_; j += 32) sm += __expf(zr[j] + v[j] - mx);
    for (int s = 16; s; s >>= 1) sm += __shfl_xor(sm, s);
    float lse = mx + __logf(sm);
    float norm = -__logf((float)(2 * Mn_));
    float lmu = (i < Mn_) ? norm : (__logf((float)Mn_) + norm);
    if (lane == 0) u[i] = lmu - lse;
}

__global__ void k_ot_col(const float* __restrict__ Z, const float* __restrict__ u,
                         float* __restrict__ v)
{
    int j = blockIdx.x;      // 0..512
    int lane = threadIdx.x;
    float mx = -3.4e38f;
    for (int i = lane; i <= Mn_; i += 32) mx = fmaxf(mx, Z[(long)i * (Mn_ + 1) + j] + u[i]);
    for (int s = 16; s; s >>= 1) mx = fmaxf(mx, __shfl_xor(mx, s));
    float sm = 0.f;
    for (int i = lane; i <= Mn_; i += 32) sm += __expf(Z[(long)i * (Mn_ + 1) + j] + u[i] - mx);
    for (int s = 16; s; s >>= 1) sm += __shfl_xor(sm, s);
    float lse = mx + __logf(sm);
    float norm = -__logf((float)(2 * Mn_));
    float lnu = (j < Mn_) ? norm : (__logf((float)Mn_) + norm);
    if (lane == 0) v[j] = lnu - lse;
}

__global__ void k_out_scores(const float* __restrict__ Z, const float* __restrict__ u,
                             const float* __restrict__ v, float* __restrict__ out)
{
    long i = blockIdx.x * (long)blockDim.x + threadIdx.x;
    long total = (long)(Mn_ + 1) * (Mn_ + 1);
    if (i >= total) return;
    int j = (int)(i % (Mn_ + 1));
    int r = (int)(i / (Mn_ + 1));
    float norm = -__logf((float)(2 * Mn_));
    out[i] = Z[i] + u[r] + v[j] - norm;
}

// ---------------------------------------------------------------------------
// Local sinkhorn (rpmnet): la is (C, 65, 65); rows 0..63 / cols 0..63 updated
// ---------------------------------------------------------------------------
__global__ void k_assemble_la(const float* __restrict__ core, const float* __restrict__ sm,
                              const float* __restrict__ tm, float* __restrict__ la)
{
    long i = blockIdx.x * (long)blockDim.x + threadIdx.x;
    long total = (long)Cc_ * 65 * 65;
    if (i >= total) return;
    int m = (int)(i % 65);
    long r = i / 65;
    int n = (int)(r % 65);
    int c = (int)(r / 65);
    float v = 0.f;
    if (n < Kk_ && m < Kk_) {
        v = core[((long)c * Kk_ + n) * Kk_ + m];
        if (sm[c * Kk_ + n] > 0.f || tm[c * Kk_ + m] > 0.f) v = -1000000.f;
    }
    la[i] = v;
}

__global__ void k_rpm_row(float* __restrict__ la)
{
    int idx = blockIdx.x;           // C*64 rows (skip bin row)
    int c = idx >> 6, r = idx & 63;
    int lane = threadIdx.x;
    float* p = la + (long)c * 65 * 65 + (long)r * 65;
    float mx = -3.4e38f;
    for (int j = lane; j < 65; j += 32) mx = fmaxf(mx, p[j]);
    for (int s = 16; s; s >>= 1) mx = fmaxf(mx, __shfl_xor(mx, s));
    float sm = 0.f;
    for (int j = lane; j < 65; j += 32) sm += __expf(p[j] - mx);
    for (int s = 16; s; s >>= 1) sm += __shfl_xor(sm, s);
    float lse = mx + __logf(sm);
    for (int j = lane; j < 65; j += 32) p[j] -= lse;
}

__global__ void k_rpm_col(float* __restrict__ la)
{
    int idx = blockIdx.x;           // C*64 cols (skip bin col)
    int c = idx >> 6, q = idx & 63;
    int lane = threadIdx.x;
    float* p = la + (long)c * 65 * 65 + q;
    float mx = -3.4e38f;
    for (int j = lane; j < 65; j += 32) mx = fmaxf(mx, p[(long)j * 65]);
    for (int s = 16; s; s >>= 1) mx = fmaxf(mx, __shfl_xor(mx, s));
    float sm = 0.f;
    for (int j = lane; j < 65; j += 32) sm += __expf(p[(long)j * 65] - mx);
    for (int s = 16; s; s >>= 1) sm += __shfl_xor(sm, s);
    float lse = mx + __logf(sm);
    for (int j = lane; j < 65; j += 32) p[(long)j * 65] -= lse;
}

// ---------------------------------------------------------------------------
// gt: per patch-pair c, 64x64 correspondence + slack row/col + masking
// blockDim = 64
// ---------------------------------------------------------------------------
__global__ void k_gt(const float* __restrict__ sc, const float* __restrict__ tc,
                     const float* __restrict__ sm, const float* __restrict__ tm,
                     float* __restrict__ gt)
{
    int c = blockIdx.x;
    int t = threadIdx.x;            // 0..63
    __shared__ float tl[Kk_ * 3];
    __shared__ float sl[Kk_ * 3];
    for (int j = t; j < Kk_ * 3; j += 64) { tl[j] = tc[c * Kk_ * 3 + j]; sl[j] = sc[c * Kk_ * 3 + j]; }
    __syncthreads();
    float px = sl[t * 3], py = sl[t * 3 + 1], pz = sl[t * 3 + 2];
    float* g = gt + (long)c * 65 * 65;
    float rowsum = 0.f;
    for (int m2 = 0; m2 < Kk_; ++m2) {
        float dx = px - tl[m2 * 3], dy = py - tl[m2 * 3 + 1], dz = pz - tl[m2 * 3 + 2];
        float d2 = fmaxf(dx * dx + dy * dy + dz * dz, 0.f);
        float val = (sqrtf(d2) < POS_MARGIN_) ? 1.f : 0.f;
        g[t * 65 + m2] = val;
        rowsum += val;
    }
    __syncthreads();
    float colsum = 0.f;
    for (int n2 = 0; n2 < Kk_; ++n2) colsum += g[n2 * 65 + t];
    __syncthreads();
    float smv = sm[c * Kk_ + t], tmv = tm[c * Kk_ + t];
    for (int m2 = 0; m2 < Kk_; ++m2) {
        float tmm = tm[c * Kk_ + m2];
        g[t * 65 + m2] *= (1.f - smv) * (1.f - tmm);
    }
    g[t * 65 + 64]  = fmaxf(1.f - rowsum, 0.f) * (1.f - smv);
    g[64 * 65 + t]  = fmaxf(1.f - colsum, 0.f) * (1.f - tmv);
    if (t == 0) g[64 * 65 + 64] = 0.f;
}

// ---------------------------------------------------------------------------
// Host-side orchestration
// ---------------------------------------------------------------------------
static inline void gemm(hipStream_t s,
                        const float* A, int tA, int lda, long sA,
                        const float* B, int tB, int ldb, long sB,
                        const float* bias,
                        const float* res, long sRes,
                        float* Y, long sY,
                        int O, int I, int N, int batch, float alpha, int relu)
{
    dim3 grid((unsigned)((N + 63) / 64), (unsigned)(O / 16), (unsigned)batch);
    if (tA == 0 && tB == 0)
        k_wmma_gemm<0,0><<<grid, 128, 0, s>>>(A, lda, sA, B, ldb, sB, bias, res, sRes, Y, sY, O, I, N, alpha, relu);
    else if (tA == 0 && tB == 1)
        k_wmma_gemm<0,1><<<grid, 128, 0, s>>>(A, lda, sA, B, ldb, sB, bias, res, sRes, Y, sY, O, I, N, alpha, relu);
    else if (tA == 1 && tB == 0)
        k_wmma_gemm<1,0><<<grid, 128, 0, s>>>(A, lda, sA, B, ldb, sB, bias, res, sRes, Y, sY, O, I, N, alpha, relu);
    else
        k_wmma_gemm<1,1><<<grid, 128, 0, s>>>(A, lda, sA, B, ldb, sB, bias, res, sRes, Y, sY, O, I, N, alpha, relu);
}

struct AttnBufs { float *q, *k, *v, *att, *msg, *msg2, *cat, *h1; };

static void attn_block_launch(hipStream_t s, int B, int Dm, int N, int Nm,
                              const float* x, const float* src, float* y,
                              const float* wq, const float* bq,
                              const float* wk, const float* bk,
                              const float* wv, const float* bv,
                              const float* wm, const float* bm,
                              const float* w1, const float* b1,
                              const float* w2, const float* b2,
                              const AttnBufs& wb)
{
    int Dh = Dm / Hh_;
    long sX = (long)Dm * N, sSrc = (long)Dm * Nm;
    // q,k,v
    gemm(s, wq, 0, Dm, 0, x,   0, N,  sX,   bq, nullptr, 0, wb.q, sX,   Dm, Dm, N,  B, 1.f, 0);
    gemm(s, wk, 0, Dm, 0, src, 0, Nm, sSrc, bk, nullptr, 0, wb.k, sSrc, Dm, Dm, Nm, B, 1.f, 0);
    gemm(s, wv, 0, Dm, 0, src, 0, Nm, sSrc, bv, nullptr, 0, wb.v, sSrc, Dm, Dm, Nm, B, 1.f, 0);
    // attention logits: att[bh, n, m] = (1/sqrt(Dh)) * sum_d q[d,n] k[d,m]
    int BH = B * Hh_;
    gemm(s, wb.q, 1, N, (long)Dh * N, wb.k, 0, Nm, (long)Dh * Nm,
         nullptr, nullptr, 0, wb.att, (long)N * Nm, N, Dh, Nm, BH, 1.f / sqrtf((float)Dh), 0);
    k_softmax_rows<<<(unsigned)(BH * N), 32, 0, s>>>(wb.att, Nm);
    // msg[bh, d, n] = sum_m v[d,m] att[n,m]
    gemm(s, wb.v, 0, Nm, (long)Dh * Nm, wb.att, 1, Nm, (long)N * Nm,
         nullptr, nullptr, 0, wb.msg, (long)Dh * N, Dh, Nm, N, BH, 1.f, 0);
    // msg2 = wm @ msg + bm
    gemm(s, wm, 0, Dm, 0, wb.msg, 0, N, sX, bm, nullptr, 0, wb.msg2, sX, Dm, Dm, N, B, 1.f, 0);
    // cat = [x; msg2]
    long tot = (long)B * 2 * Dm * N;
    k_concat<<<(unsigned)((tot + 255) / 256), 256, 0, s>>>(x, wb.msg2, wb.cat, B, Dm, N);
    // h1 = relu(w1 @ cat + b1)
    gemm(s, w1, 0, 2 * Dm, 0, wb.cat, 0, N, (long)2 * Dm * N,
         b1, nullptr, 0, wb.h1, (long)2 * Dm * N, 2 * Dm, 2 * Dm, N, B, 1.f, 1);
    // y = x + w2 @ h1 + b2
    gemm(s, w2, 0, 2 * Dm, 0, wb.h1, 0, N, (long)2 * Dm * N,
         b2, x, sX, y, sX, Dm, 2 * Dm, N, B, 1.f, 0);
}

extern "C" void kernel_launch(void* const* d_in, const int* in_sizes, int n_in,
                              void* d_out, int out_size, void* d_ws, size_t ws_size,
                              hipStream_t stream)
{
    (void)in_sizes; (void)n_in; (void)out_size; (void)ws_size;

    // ---- inputs (setup_inputs dict order) ----
    const float* src_pcd_c   = (const float*)d_in[0];
    const float* tgt_pcd_c   = (const float*)d_in[1];
    const float* src_node_c  = (const float*)d_in[2];
    const float* tgt_node_c  = (const float*)d_in[3];
    const float* src_feats_c = (const float*)d_in[4];
    const float* tgt_feats_c = (const float*)d_in[5];
    const float* src_final_f = (const float*)d_in[6];
    const float* tgt_final_f = (const float*)d_in[7];
    const float* rot         = (const float*)d_in[8];
    const float* trans       = (const float*)d_in[9];
    const float* src_pcd_raw = (const float*)d_in[10];
    const float* tgt_pcd_raw = (const float*)d_in[11];
    const int*   node_corr   = (const int*)d_in[12];
    const float* c_wq = (const float*)d_in[13]; const float* c_bq = (const float*)d_in[14];
    const float* c_wk = (const float*)d_in[15]; const float* c_bk = (const float*)d_in[16];
    const float* c_wv = (const float*)d_in[17]; const float* c_bv = (const float*)d_in[18];
    const float* c_wm = (const float*)d_in[19]; const float* c_bm = (const float*)d_in[20];
    const float* c_w1 = (const float*)d_in[21]; const float* c_b1 = (const float*)d_in[22];
    const float* c_w2 = (const float*)d_in[23]; const float* c_b2 = (const float*)d_in[24];
    const float* l_wq = (const float*)d_in[25]; const float* l_bq = (const float*)d_in[26];
    const float* l_wk = (const float*)d_in[27]; const float* l_bk = (const float*)d_in[28];
    const float* l_wv = (const float*)d_in[29]; const float* l_bv = (const float*)d_in[30];
    const float* l_wm = (const float*)d_in[31]; const float* l_bm = (const float*)d_in[32];
    const float* l_w1 = (const float*)d_in[33]; const float* l_b1 = (const float*)d_in[34];
    const float* l_w2 = (const float*)d_in[35]; const float* l_b2 = (const float*)d_in[36];
    const float* fp_w  = (const float*)d_in[37]; const float* fp_b  = (const float*)d_in[38];
    const float* lfp_w = (const float*)d_in[39]; const float* lfp_b = (const float*)d_in[40];
    const float* bin_score = (const float*)d_in[41];

    float* out_scores = (float*)d_out;                                 // 513*513
    float* out_ls     = out_scores + (long)(Mn_ + 1) * (Mn_ + 1);      // 256*65*65
    float* out_gt     = out_ls + (long)Cc_ * 65 * 65;                  // 256*65*65

    // ---- workspace bump allocator ----
    char* wsp = (char*)d_ws;
    size_t off = 0;
    auto alloc = [&](size_t bytes) -> void* {
        void* p = wsp + off;
        off = (off + bytes + 255) & ~(size_t)255;
        return p;
    };
    const size_t FEAT = (size_t)Cc_ * DLc_ * Kk_;        // 2097152 (>= Dc_*Mn_)
    const size_t BIG  = (size_t)Cc_ * 2 * DLc_ * Kk_;    // 4194304 (>= 2*Dc_*Mn_, H*512*512)

    float* fX1 = (float*)alloc(FEAT * 4); float* fX2 = (float*)alloc(FEAT * 4);
    float* fX3 = (float*)alloc(FEAT * 4); float* fX4 = (float*)alloc(FEAT * 4);
    float* fQ  = (float*)alloc(FEAT * 4); float* fK  = (float*)alloc(FEAT * 4);
    float* fV  = (float*)alloc(FEAT * 4);
    float* fMSG  = (float*)alloc(FEAT * 4); float* fMSG2 = (float*)alloc(FEAT * 4);
    float* fATT  = (float*)alloc(BIG * 4);
    float* fCAT  = (float*)alloc(BIG * 4);  float* fH1 = (float*)alloc(BIG * 4);
    float* fZ  = (float*)alloc((size_t)(Mn_ + 1) * (Mn_ + 1) * 4);
    float* fU  = (float*)alloc((Mn_ + 1) * 4);
    float* fVv = (float*)alloc((Mn_ + 1) * 4);
    float* fLA = (float*)alloc((size_t)Cc_ * 65 * 65 * 4);
    float* fMASK_S = (float*)alloc((size_t)Mn_ * Kk_ * 4);
    float* fMASK_T = (float*)alloc((size_t)Mn_ * Kk_ * 4);
    float* fSELM_S = (float*)alloc((size_t)Cc_ * Kk_ * 4);
    float* fSELM_T = (float*)alloc((size_t)Cc_ * Kk_ * 4);
    float* fSC = (float*)alloc((size_t)Cc_ * Kk_ * 3 * 4);
    float* fTC = (float*)alloc((size_t)Cc_ * Kk_ * 3 * 4);
    int* iIDS_S  = (int*)alloc((size_t)NS_ * 4);
    int* iIDS_T  = (int*)alloc((size_t)NT_ * 4);
    int* iCNT_S  = (int*)alloc(Mn_ * 4);
    int* iCNT_T  = (int*)alloc(Mn_ * 4);
    int* iPATCH_S = (int*)alloc((size_t)Mn_ * Kk_ * 4);
    int* iPATCH_T = (int*)alloc((size_t)Mn_ * Kk_ * 4);
    int* iSEL_S  = (int*)alloc((size_t)Cc_ * Kk_ * 4);
    int* iSEL_T  = (int*)alloc((size_t)Cc_ * Kk_ * 4);

    AttnBufs wb = { fQ, fK, fV, fATT, fMSG, fMSG2, fCAT, fH1 };

    // =====================================================================
    // Phase 1: coarse attention + OT scores
    // =====================================================================
    hipMemcpyAsync(fX1, src_feats_c, (size_t)Dc_ * Mn_ * 4, hipMemcpyDeviceToDevice, stream);
    hipMemcpyAsync(fX2, tgt_feats_c, (size_t)Dc_ * Mn_ * 4, hipMemcpyDeviceToDevice, stream);

    auto cW = [&](const float* p, int i, long sz) { return p + (long)i * sz; };
    for (int L = 0; L < 3; ++L) {
        const float* wq = cW(c_wq, L, (long)Dc_ * Dc_);     const float* bq = cW(c_bq, L, Dc_);
        const float* wk = cW(c_wk, L, (long)Dc_ * Dc_);     const float* bk = cW(c_bk, L, Dc_);
        const float* wv = cW(c_wv, L, (long)Dc_ * Dc_);     const float* bv = cW(c_bv, L, Dc_);
        const float* wm = cW(c_wm, L, (long)Dc_ * Dc_);     const float* bm = cW(c_bm, L, Dc_);
        const float* w1 = cW(c_w1, L, (long)2 * Dc_ * 2 * Dc_); const float* b1 = cW(c_b1, L, 2 * Dc_);
        const float* w2 = cW(c_w2, L, (long)Dc_ * 2 * Dc_); const float* b2 = cW(c_b2, L, Dc_);
        if (L == 1) {
            // cross-attention; tf update sees updated sf (reference semantics)
            attn_block_launch(stream, 1, Dc_, Mn_, Mn_, fX3, fX4, fX1, wq, bq, wk, bk, wv, bv, wm, bm, w1, b1, w2, b2, wb);
            attn_block_launch(stream, 1, Dc_, Mn_, Mn_, fX4, fX1, fX2, wq, bq, wk, bk, wv, bv, wm, bm, w1, b1, w2, b2, wb);
        } else {
            const float* xs = fX1; const float* xt = fX2;
            attn_block_launch(stream, 1, Dc_, Mn_, Mn_, xs, xs, fX3, wq, bq, wk, bk, wv, bv, wm, bm, w1, b1, w2, b2, wb);
            attn_block_launch(stream, 1, Dc_, Mn_, Mn_, xt, xt, fX4, wq, bq, wk, bk, wv, bv, wm, bm, w1, b1, w2, b2, wb);
        }
    }
    // final projection (results currently in fX3/fX4 after layer 2)
    gemm(stream, fp_w, 0, Dc_, 0, fX3, 0, Mn_, 0, fp_b, nullptr, 0, fQ, 0, Dc_, Dc_, Mn_, 1, 1.f, 0);
    gemm(stream, fp_w, 0, Dc_, 0, fX4, 0, Mn_, 0, fp_b, nullptr, 0, fK, 0, Dc_, Dc_, Mn_, 1, 1.f, 0);
    // scores core: (1/sqrt(D)) * sf^T tf  -> fATT (512x512)
    gemm(stream, fQ, 1, Mn_, 0, fK, 0, Mn_, 0, nullptr, nullptr, 0, fATT, 0,
         Mn_, Dc_, Mn_, 1, 1.f / sqrtf((float)Dc_), 0);
    {
        long tot = (long)(Mn_ + 1) * (Mn_ + 1);
        k_assemble_Z<<<(unsigned)((tot + 255) / 256), 256, 0, stream>>>(fATT, bin_score, fZ);
        k_zero_f32<<<3, 256, 0, stream>>>(fU, Mn_ + 1);
        k_zero_f32<<<3, 256, 0, stream>>>(fVv, Mn_ + 1);
        for (int it = 0; it < ITERS_; ++it) {
            k_ot_row<<<Mn_ + 1, 32, 0, stream>>>(fZ, fU, fVv);
            k_ot_col<<<Mn_ + 1, 32, 0, stream>>>(fZ, fU, fVv);
        }
        k_out_scores<<<(unsigned)((tot + 255) / 256), 256, 0, stream>>>(fZ, fU, fVv, out_scores);
    }

    // =====================================================================
    // Phase 2: geometry (nn ids, patches, gather, gt)
    // =====================================================================
    k_nn_argmin<<<(NS_ + 255) / 256, 256, 0, stream>>>(src_pcd_c, NS_, src_node_c, Mn_, iIDS_S);
    k_nn_argmin<<<(NT_ + 255) / 256, 256, 0, stream>>>(tgt_pcd_c, NT_, tgt_node_c, Mn_, iIDS_T);
    k_zero_i32<<<2, 256, 0, stream>>>(iCNT_S, Mn_);
    k_zero_i32<<<2, 256, 0, stream>>>(iCNT_T, Mn_);
    k_build_patch<<<(NS_ + 255) / 256, 256, 0, stream>>>(iIDS_S, NS_, iCNT_S, iPATCH_S);
    k_build_patch<<<(NT_ + 255) / 256, 256, 0, stream>>>(iIDS_T, NT_, iCNT_T, iPATCH_T);
    k_patch_mask<<<(Mn_ * Kk_ + 255) / 256, 256, 0, stream>>>(iCNT_S, iPATCH_S, fMASK_S);
    k_patch_mask<<<(Mn_ * Kk_ + 255) / 256, 256, 0, stream>>>(iCNT_T, iPATCH_T, fMASK_T);
    k_make_sel<<<(Cc_ * Kk_ + 255) / 256, 256, 0, stream>>>(iPATCH_S, fMASK_S, node_corr, 0, iSEL_S, fSELM_S);
    k_make_sel<<<(Cc_ * Kk_ + 255) / 256, 256, 0, stream>>>(iPATCH_T, fMASK_T, node_corr, 1, iSEL_T, fSELM_T);
    k_gather_pts<<<(Cc_ * Kk_ + 255) / 256, 256, 0, stream>>>(src_pcd_raw, iSEL_S, rot, trans, Cc_ * Kk_, fSC);
    k_gather_pts<<<(Cc_ * Kk_ + 255) / 256, 256, 0, stream>>>(tgt_pcd_raw, iSEL_T, nullptr, nullptr, Cc_ * Kk_, fTC);
    k_gt<<<Cc_, 64, 0, stream>>>(fSC, fTC, fSELM_S, fSELM_T, out_gt);

    // =====================================================================
    // Phase 3: local attention + rpmnet sinkhorn
    // =====================================================================
    {
        long tot = (long)Cc_ * DLc_ * Kk_;
        k_gather_feats<<<(unsigned)((tot + 255) / 256), 256, 0, stream>>>(src_final_f, iSEL_S, fX1);
        k_gather_feats<<<(unsigned)((tot + 255) / 256), 256, 0, stream>>>(tgt_final_f, iSEL_T, fX2);
    }
    for (int L = 0; L < 3; ++L) {
        const float* wq = cW(l_wq, L, (long)DLc_ * DLc_);      const float* bq = cW(l_bq, L, DLc_);
        const float* wk = cW(l_wk, L, (long)DLc_ * DLc_);      const float* bk = cW(l_bk, L, DLc_);
        const float* wv = cW(l_wv, L, (long)DLc_ * DLc_);      const float* bv = cW(l_bv, L, DLc_);
        const float* wm = cW(l_wm, L, (long)DLc_ * DLc_);      const float* bm = cW(l_bm, L, DLc_);
        const float* w1 = cW(l_w1, L, (long)2 * DLc_ * 2 * DLc_); const float* b1 = cW(l_b1, L, 2 * DLc_);
        const float* w2 = cW(l_w2, L, (long)DLc_ * 2 * DLc_);  const float* b2 = cW(l_b2, L, DLc_);
        if (L == 1) {
            attn_block_launch(stream, Cc_, DLc_, Kk_, Kk_, fX3, fX4, fX1, wq, bq, wk, bk, wv, bv, wm, bm, w1, b1, w2, b2, wb);
            attn_block_launch(stream, Cc_, DLc_, Kk_, Kk_, fX4, fX1, fX2, wq, bq, wk, bk, wv, bv, wm, bm, w1, b1, w2, b2, wb);
        } else {
            attn_block_launch(stream, Cc_, DLc_, Kk_, Kk_, fX1, fX1, fX3, wq, bq, wk, bk, wv, bv, wm, bm, w1, b1, w2, b2, wb);
            attn_block_launch(stream, Cc_, DLc_, Kk_, Kk_, fX2, fX2, fX4, wq, bq, wk, bk, wv, bv, wm, bm, w1, b1, w2, b2, wb);
        }
    }
    // local final projection (from fX3/fX4) then score core
    gemm(stream, lfp_w, 0, DLc_, 0, fX3, 0, Kk_, (long)DLc_ * Kk_, lfp_b, nullptr, 0,
         fQ, (long)DLc_ * Kk_, DLc_, DLc_, Kk_, Cc_, 1.f, 0);
    gemm(stream, lfp_w, 0, DLc_, 0, fX4, 0, Kk_, (long)DLc_ * Kk_, lfp_b, nullptr, 0,
         fK, (long)DLc_ * Kk_, DLc_, DLc_, Kk_, Cc_, 1.f, 0);
    gemm(stream, fQ, 1, Kk_, (long)DLc_ * Kk_, fK, 0, Kk_, (long)DLc_ * Kk_,
         nullptr, nullptr, 0, fATT, (long)Kk_ * Kk_,
         Kk_, DLc_, Kk_, Cc_, 1.f / sqrtf((float)DLc_), 0);
    {
        long tot = (long)Cc_ * 65 * 65;
        k_assemble_la<<<(unsigned)((tot + 255) / 256), 256, 0, stream>>>(fATT, fSELM_S, fSELM_T, fLA);
        for (int it = 0; it < ITERS_; ++it) {
            k_rpm_row<<<Cc_ * Kk_, 32, 0, stream>>>(fLA);
            k_rpm_col<<<Cc_ * Kk_, 32, 0, stream>>>(fLA);
        }
        hipMemcpyAsync(out_ls, fLA, (size_t)Cc_ * 65 * 65 * 4, hipMemcpyDeviceToDevice, stream);
    }
}
